// PCF_net_40621800685955
// MI455X (gfx1250) — compile-verified, matrix operand order checked by reading, not run on hardware
//
#include <hip/hip_runtime.h>
#include <hip/hip_bf16.h>

// ---------------------------------------------------------------------------
// PCF_net forward for MI455X (gfx1250, wave32, WMMA).
// All GEMM-like ops go through an LDS-staged WMMA f16->f32 GEMM kernel with
// fused BN + activation + alpha-scale + residual epilogue, plus next-tile
// global_prefetch to overlap HBM latency with LDS/WMMA work.
// ---------------------------------------------------------------------------

typedef __attribute__((ext_vector_type(16))) _Float16 v16h;
typedef __attribute__((ext_vector_type(8)))  _Float16 v8h;
typedef __attribute__((ext_vector_type(2)))  _Float16 v2h;
typedef __attribute__((ext_vector_type(8)))  float    v8f;

#define NPT    4096
#define BATCH  4
#define KNN    20
#define CH     1408
#define PDIM   512
#define NKCOL  (NPT * KNN)
#define BNEPS  1e-5f
#define LEAK   0.2f
#define LDSTR  48          // LDS row stride in halves (96B, 16B-aligned, bank-spread)

// ===========================================================================
// Generic batched GEMM:  C[b] = epilogue( A[b](MxK) * B[b](KxN) )
//  TA=0 -> A[m*lda+k], TA=1 -> A[k*lda+m]
//  TB=0 -> B[k*ldb+n], TB=1 -> B[n*ldb+k]
//  epilogue: BN (per-M-row affine) -> act (0 none / 1 relu / 2 leaky 0.2)
//            -> *alphaPtr[0] -> + R[b][m*ldc+n]
// Block = 256 thr = 8 waves arranged 4(M) x 2(N) of 16x16 WMMA tiles -> 64x32.
// Tiles staged in LDS as f16 (gather phase fully precedes store phase so the
// global loads batch under one wait); fragments read as 2x ds_load_b128.
// ===========================================================================
template <int TA, int TB>
__global__ void gemm_wmma_kernel(
    const float* __restrict__ A, int lda, long long sA_,
    const float* __restrict__ B, int ldb, long long sB_,
    float* __restrict__ C, int ldc, long long sC_,
    int M, int N, int K,
    const float* __restrict__ bnG, const float* __restrict__ bnB,
    const float* __restrict__ bnM, const float* __restrict__ bnV,
    int act, const float* __restrict__ alphaPtr,
    const float* __restrict__ R, long long sR)
{
  __shared__ _Float16 sA[64 * LDSTR];    // [rowLocal][k]   (row-major)
  __shared__ _Float16 sB[32 * LDSTR];    // [colLocal][k]   (transposed)

  const int b = blockIdx.z;
  const float* Ab = A + (long long)b * sA_;
  const float* Bb = B + (long long)b * sB_;
  float*       Cb = C + (long long)b * sC_;

  const int tid   = threadIdx.x;
  const int wave  = tid >> 5;
  const int lane  = tid & 31;
  const int wm    = wave & 3;
  const int wn    = wave >> 2;
  const int tileM = blockIdx.y * 64;
  const int tileN = blockIdx.x * 32;

  const int lhalf = lane >> 4;           // selects K-subrange per ISA layout
  const int l16   = lane & 15;
  const int kbase = lhalf * 8;

  v8f acc = {0.f, 0.f, 0.f, 0.f, 0.f, 0.f, 0.f, 0.f};

  for (int k0 = 0; k0 < K; k0 += 32) {
    // ================= gather phase: ALL global loads, no LDS ops ==========
    // A tile (64x32) as 1024 pairs, 4 pairs/thread.
    //  TA=0: pair = (k,k+1) same row  -> contiguous in mem AND in LDS
    //  TA=1: pair = (row,row+1) same k -> contiguous in mem, 2 LDS b16 stores
    float ax0[4], ax1[4];
    int   arow[4], akk[4];
#pragma unroll
    for (int j = 0; j < 4; ++j) {
      int p = j * 256 + tid;
      int row, kk;
      if (TA) { kk = p >> 5; row = (p & 31) * 2; }
      else    { row = p >> 4; kk = (p & 15) * 2; }
      arow[j] = row; akk[j] = kk;
      int gr = tileM + row, gk = k0 + kk;
      bool ok0, ok1;
      long long i0, i1;
      if (TA) {
        ok0 = (gr < M)     && (gk < K);
        ok1 = (gr + 1 < M) && (gk < K);
        i0  = (long long)gk * lda + gr;
        i1  = i0 + 1;
      } else {
        ok0 = (gr < M) && (gk < K);
        ok1 = (gr < M) && (gk + 1 < K);
        i0  = (long long)gr * lda + gk;
        i1  = i0 + 1;
      }
      float x0 = Ab[ok0 ? i0 : 0];
      float x1 = Ab[ok1 ? i1 : 0];
      ax0[j] = ok0 ? x0 : 0.f;
      ax1[j] = ok1 ? x1 : 0.f;
      // speculative prefetch of the next K-tile (gfx1250 global_prefetch)
      int gkN = gk + 32;
      if (gkN < K && gr < M) {
        long long ip = TA ? (long long)gkN * lda + gr : (long long)gr * lda + gkN;
        __builtin_prefetch(&Ab[ip], 0, 3);
      }
    }
    // B tile (32x32) as 512 pairs, 2 pairs/thread; pair = (k,k+1) same col
    //  -> always contiguous in LDS [col][k]; contiguous in mem iff TB=1.
    float bx0[2], bx1[2];
    int   bcol[2], bkk[2];
#pragma unroll
    for (int j = 0; j < 2; ++j) {
      int p = j * 256 + tid;
      int col, kk;
      if (TB) { col = p >> 4; kk = (p & 15) * 2; }
      else    { col = p & 31; kk = (p >> 5) * 2; }
      bcol[j] = col; bkk[j] = kk;
      int gc = tileN + col, gk = k0 + kk;
      bool ok0 = (gc < N) && (gk < K);
      bool ok1 = (gc < N) && (gk + 1 < K);
      long long i0, i1;
      if (TB) { i0 = (long long)gc * ldb + gk; i1 = i0 + 1; }
      else    { i0 = (long long)gk * ldb + gc; i1 = i0 + ldb; }
      float x0 = Bb[ok0 ? i0 : 0];
      float x1 = Bb[ok1 ? i1 : 0];
      bx0[j] = ok0 ? x0 : 0.f;
      bx1[j] = ok1 ? x1 : 0.f;
      int gkN = gk + 32;
      if (gkN < K && gc < N) {
        long long ip = TB ? (long long)gc * ldb + gkN : (long long)gkN * ldb + gc;
        __builtin_prefetch(&Bb[ip], 0, 3);
      }
    }

    __syncthreads();   // previous iteration's fragment reads complete
    // ================= store phase: packed f16 stores to LDS ===============
#pragma unroll
    for (int j = 0; j < 4; ++j) {
      if (TA) {
        sA[arow[j] * LDSTR + akk[j]]       = (_Float16)ax0[j];
        sA[(arow[j] + 1) * LDSTR + akk[j]] = (_Float16)ax1[j];
      } else {
        v2h pk = { (_Float16)ax0[j], (_Float16)ax1[j] };
        *(v2h*)&sA[arow[j] * LDSTR + akk[j]] = pk;      // kk even -> 4B aligned
      }
    }
#pragma unroll
    for (int j = 0; j < 2; ++j) {
      v2h pk = { (_Float16)bx0[j], (_Float16)bx1[j] };
      *(v2h*)&sB[bcol[j] * LDSTR + bkk[j]] = pk;
    }
    __syncthreads();

    // ---- fragments: 2x b128 LDS loads each (16B aligned: kbase*2 in {0,16})
    const _Float16* pa = &sA[(wm * 16 + l16) * LDSTR + kbase];
    const _Float16* pb = &sB[(wn * 16 + l16) * LDSTR + kbase];
    v8h a0 = *(const v8h*)pa;
    v8h a1 = *(const v8h*)(pa + 16);
    v8h b0 = *(const v8h*)pb;
    v8h b1 = *(const v8h*)(pb + 16);
    v16h af = __builtin_shufflevector(a0, a1, 0,1,2,3,4,5,6,7,8,9,10,11,12,13,14,15);
    v16h bf = __builtin_shufflevector(b0, b1, 0,1,2,3,4,5,6,7,8,9,10,11,12,13,14,15);

    acc = __builtin_amdgcn_wmma_f32_16x16x32_f16(false, af, false, bf,
                                                 (short)0, acc, false, false);
  }

  const float aScale = alphaPtr ? alphaPtr[0] : 1.0f;
#pragma unroll
  for (int r = 0; r < 8; ++r) {
    int m = tileM + wm * 16 + lhalf * 8 + r;   // C/D: VGPR r, lane-half -> M+8
    int n = tileN + wn * 16 + l16;
    if (m < M && n < N) {
      float v = acc[r];
      if (bnG) {
        float s = bnG[m] * rsqrtf(bnV[m] + BNEPS);
        v = (v - bnM[m]) * s + bnB[m];
      }
      if (act == 1)      v = v > 0.f ? v : 0.f;
      else if (act == 2) v = v > 0.f ? v : LEAK * v;
      v *= aScale;
      if (R) v += R[(long long)b * sR + (long long)m * ldc + n];
      Cb[(long long)m * ldc + n] = v;
    }
  }
}

// ===========================================================================
// Elementwise / irregular kernels
// ===========================================================================
__global__ void split_kernel(const float* __restrict__ x,
                             float* __restrict__ COR, float* __restrict__ RGB)
{
  int t = blockIdx.x * blockDim.x + threadIdx.x;
  int b = blockIdx.y;
  if (t >= NPT) return;
  const float* xb = x + (long long)b * 9 * NPT;
  float* cb = COR + (long long)b * 6 * NPT;
  float* rb = RGB + (long long)b * 3 * NPT;
#pragma unroll
  for (int c = 0; c < 3; ++c) {
    cb[c * NPT + t]       = xb[c * NPT + t];         // x[:, :3]
    cb[(3 + c) * NPT + t] = xb[(6 + c) * NPT + t];   // x[:, 6:9]
    rb[c * NPT + t]       = xb[(3 + c) * NPT + t];   // x[:, 3:6]
  }
}

__global__ void sqnorm_kernel(const float* __restrict__ X, long long sX, int C,
                              float* __restrict__ nrm)
{
  int t = blockIdx.x * blockDim.x + threadIdx.x;
  int b = blockIdx.y;
  if (t >= NPT) return;
  const float* Xb = X + (long long)b * sX;
  float s = 0.f;
  for (int c = 0; c < C; ++c) { float v = Xb[(long long)c * NPT + t]; s += v * v; }
  nrm[(long long)b * NPT + t] = s;
}

__global__ void knn_kernel(const float* __restrict__ X, long long sX, int C,
                           const float* __restrict__ nrm, int* __restrict__ idx)
{
  int t = blockIdx.x * blockDim.x + threadIdx.x;
  int b = blockIdx.y;
  if (t >= NPT) return;
  const float* Xb = X + (long long)b * sX;
  const float* nb = nrm + (long long)b * NPT;
  float qv[134];
  for (int c = 0; c < C; ++c) qv[c] = Xb[(long long)c * NPT + t];
  float best[KNN]; int bi[KNN];
#pragma unroll
  for (int i = 0; i < KNN; ++i) { best[i] = -3.4e38f; bi[i] = 0; }
  const float qq = nb[t];
  for (int m = 0; m < NPT; ++m) {
    float dot = 0.f;
    for (int c = 0; c < C; ++c) dot += qv[c] * Xb[(long long)c * NPT + m];
    float s = 2.f * dot - qq - nb[m];           // == pair score; top-k largest
    if (s > best[KNN - 1]) {
      int p = KNN - 1;
      while (p > 0 && best[p - 1] < s) { best[p] = best[p - 1]; bi[p] = bi[p - 1]; --p; }
      best[p] = s; bi[p] = m;
    }
  }
  int* ob = idx + ((long long)b * NPT + t) * KNN;
#pragma unroll
  for (int i = 0; i < KNN; ++i) ob[i] = bi[i];
}

// F[(2C) x (N*k)] : first C rows = neigh-center, next C rows = center
__global__ void graphfeat_kernel(const float* __restrict__ Xb, int C,
                                 const int* __restrict__ idxb,
                                 float* __restrict__ Fo)
{
  long long t   = blockIdx.x * (long long)blockDim.x + threadIdx.x;
  long long tot = 2ll * C * NKCOL;
  if (t >= tot) return;
  int ch        = (int)(t / NKCOL);
  long long col = t - (long long)ch * NKCOL;
  int n = (int)(col / KNN);
  int j = (int)(col - (long long)n * KNN);
  float v;
  if (ch < C) {
    int nb = idxb[(long long)n * KNN + j];
    v = Xb[(long long)ch * NPT + nb] - Xb[(long long)ch * NPT + n];
  } else {
    v = Xb[(long long)(ch - C) * NPT + n];
  }
  Fo[t] = v;
}

__global__ void maxk_kernel(const float* __restrict__ E, float* __restrict__ out, int Cc)
{
  int t = blockIdx.x * blockDim.x + threadIdx.x;
  int tot = Cc * NPT;
  if (t >= tot) return;
  int ch = t / NPT, n = t - ch * NPT;
  const float* e = E + (long long)ch * NKCOL + (long long)n * KNN;
  float m = e[0];
#pragma unroll
  for (int j = 1; j < KNN; ++j) m = fmaxf(m, e[j]);
  out[(long long)ch * NPT + n] = m;
}

__global__ void maxbcast_kernel(const float* __restrict__ G, long long sG,
                                float* __restrict__ H, long long sH)
{
  int ch = blockIdx.x, b = blockIdx.y;
  const float* g = G + (long long)b * sG + (long long)ch * NPT;
  __shared__ float red[256];
  float m = -3.4e38f;
  for (int n = threadIdx.x; n < NPT; n += blockDim.x) m = fmaxf(m, g[n]);
  red[threadIdx.x] = m; __syncthreads();
  for (int s = blockDim.x >> 1; s > 0; s >>= 1) {
    if (threadIdx.x < (unsigned)s) red[threadIdx.x] = fmaxf(red[threadIdx.x], red[threadIdx.x + s]);
    __syncthreads();
  }
  float mv = red[0];
  float* h = H + (long long)b * sH + (long long)ch * NPT;
  for (int n = threadIdx.x; n < NPT; n += blockDim.x) h[n] = mv;
}

// softmax(max_row - S) over last dim == softmax(-S); stabilized with row min
__global__ void negsoftmax_kernel(float* __restrict__ S, int D)
{
  long long row = blockIdx.x;
  float* s = S + row * (long long)D;
  __shared__ float red[256];
  float mn = 3.4e38f;
  for (int d = threadIdx.x; d < D; d += blockDim.x) mn = fminf(mn, s[d]);
  red[threadIdx.x] = mn; __syncthreads();
  for (int st = blockDim.x >> 1; st > 0; st >>= 1) {
    if (threadIdx.x < (unsigned)st) red[threadIdx.x] = fminf(red[threadIdx.x], red[threadIdx.x + st]);
    __syncthreads();
  }
  mn = red[0]; __syncthreads();
  float sum = 0.f;
  for (int d = threadIdx.x; d < D; d += blockDim.x) sum += __expf(mn - s[d]);
  red[threadIdx.x] = sum; __syncthreads();
  for (int st = blockDim.x >> 1; st > 0; st >>= 1) {
    if (threadIdx.x < (unsigned)st) red[threadIdx.x] += red[threadIdx.x + st];
    __syncthreads();
  }
  float inv = 1.f / red[0];
  for (int d = threadIdx.x; d < D; d += blockDim.x) s[d] = __expf(mn - s[d]) * inv;
}

__global__ void copy_rows_kernel(float* __restrict__ dst, long long dStride,
                                 const float* __restrict__ src, long long sStride,
                                 long long tot)
{
  long long t = blockIdx.x * (long long)blockDim.x + threadIdx.x;
  int b = blockIdx.y;
  if (t >= tot) return;
  dst[(long long)b * dStride + t] = src[(long long)b * sStride + t];
}

// ===========================================================================
// Host orchestration
// ===========================================================================
extern "C" void kernel_launch(void* const* d_in, const int* in_sizes, int n_in,
                              void* d_out, int out_size, void* d_ws, size_t ws_size,
                              hipStream_t stream)
{
  (void)in_sizes; (void)n_in; (void)out_size; (void)ws_size;
  // JAX pytree leaf order: top-level keys sorted ('k','params','x'); params keys
  // sorted; each bn dict -> (beta, gamma, mean, var).
  enum { I_K = 0, I_ALPHA = 1,
         BN1 = 2, BN1D = 6, BN2 = 10, BN2D = 14, BN3 = 18, BN3D = 22,
         BN4 = 26, BN4D = 30, BN5 = 34, BN5D = 38, BN6 = 42, BN6D = 46,
         BN7 = 50, BN8 = 54, BN9 = 58, BNC = 62, BNK = 66, BNQ = 70, BNV = 74,
         W1 = 78, W10 = 79, W1D = 80, W2 = 81, W2D = 82, W3 = 83, W3D = 84,
         W4 = 85, W4D = 86, W5 = 87, W5D = 88, W6 = 89, W6D = 90, W7 = 91,
         W8 = 92, W9 = 93, WC = 94, WK = 95, WQ = 96, WV = 97, I_X = 98 };
  auto P = [&](int i) { return (const float*)d_in[i]; };

  // ---- workspace layout (floats) ----
  float* ws = (float*)d_ws;
  size_t off = 0;
  auto alloc = [&](size_t n) { float* p = ws + off; off += (n + 255) & ~(size_t)255; return p; };

  float* h    = alloc((size_t)BATCH * CH * NPT);      // [g |inf1|inf2|inf3]
  float* COR  = alloc((size_t)BATCH * 6 * NPT);
  float* RGB  = alloc((size_t)BATCH * 3 * NPT);
  float* CC   = alloc((size_t)BATCH * 134 * NPT);     // knn concat input
  float* RC   = alloc((size_t)BATCH * 131 * NPT);     // rgb concat input
  float* T64  = alloc((size_t)BATCH * 64 * NPT);      // rgb intermediate
  float* NRM  = alloc((size_t)BATCH * NPT);
  int*   IDX  = (int*)alloc((size_t)BATCH * NPT * KNN);
  // region B: edge pipeline (F|E1|E2) OR g_full OR v (serially reused)
  float* regB = alloc((size_t)(268 + 64 + 64) * NKCOL);
  float* Fbuf = regB;
  float* E1   = regB + (size_t)268 * NKCOL;
  float* E2   = E1   + (size_t)64  * NKCOL;
  float* gful = regB;
  float* vbuf = regB;
  // region C: (q|kk|sim) OR (h7|h8|h9)
  float* regC = alloc((size_t)BATCH * (512 + 256 + 128) * NPT);
  float* qb   = regC;
  float* kkb  = qb  + (size_t)BATCH * PDIM * CH;
  float* simb = kkb + (size_t)BATCH * PDIM * CH;
  float* h7   = regC;
  float* h8   = h7 + (size_t)BATCH * 512 * NPT;
  float* h9   = h8 + (size_t)BATCH * 256 * NPT;

  const long long sH = (long long)CH * NPT;

  auto gemm = [&](const float* A, int lda, long long sA, int tA,
                  const float* B, int ldb, long long sB, int tB,
                  float* C, int ldc, long long sC,
                  int M, int N, int K, int bnBase, int act,
                  const float* alphaP, const float* R, long long sR, int batches) {
    const float *g = nullptr, *bb = nullptr, *mm = nullptr, *vv = nullptr;
    if (bnBase >= 0) { bb = P(bnBase); g = P(bnBase + 1); mm = P(bnBase + 2); vv = P(bnBase + 3); }
    dim3 grid((N + 31) / 32, (M + 63) / 64, batches);
    if (tA == 0 && tB == 0)
      gemm_wmma_kernel<0,0><<<grid, dim3(256), 0, stream>>>(A, lda, sA, B, ldb, sB,
          C, ldc, sC, M, N, K, g, bb, mm, vv, act, alphaP, R, sR);
    else if (tA == 0 && tB == 1)
      gemm_wmma_kernel<0,1><<<grid, dim3(256), 0, stream>>>(A, lda, sA, B, ldb, sB,
          C, ldc, sC, M, N, K, g, bb, mm, vv, act, alphaP, R, sR);
    else if (tA == 1 && tB == 0)
      gemm_wmma_kernel<1,0><<<grid, dim3(256), 0, stream>>>(A, lda, sA, B, ldb, sB,
          C, ldc, sC, M, N, K, g, bb, mm, vv, act, alphaP, R, sR);
    else
      gemm_wmma_kernel<1,1><<<grid, dim3(256), 0, stream>>>(A, lda, sA, B, ldb, sB,
          C, ldc, sC, M, N, K, g, bb, mm, vv, act, alphaP, R, sR);
  };
  auto copyR = [&](float* dst, long long dS, const float* src, long long sS, int rows) {
    long long tot = (long long)rows * NPT;
    dim3 grid((unsigned)((tot + 255) / 256), BATCH);
    copy_rows_kernel<<<grid, 256, 0, stream>>>(dst, dS, src, sS, tot);
  };
  auto edge_layer = [&](const float* Xc, int C, int wIdx, int bn1i, int lda1, int M1,
                        int w2Idx, int bn2i, int rowOff) {
    for (int b = 0; b < BATCH; ++b) {
      long long tot = 2ll * C * NKCOL;
      graphfeat_kernel<<<dim3((unsigned)((tot + 255) / 256)), 256, 0, stream>>>(
          Xc + (long long)b * C * NPT, C, IDX + (long long)b * NPT * KNN, Fbuf);
      gemm(P(wIdx), lda1, 0, 0, Fbuf, NKCOL, 0, 0, E1, NKCOL, 0,
           M1, NKCOL, 2 * C, bn1i, 2, nullptr, nullptr, 0, 1);
      gemm(P(w2Idx), M1, 0, 0, E1, NKCOL, 0, 0, E2, NKCOL, 0,
           64, NKCOL, M1, bn2i, 2, nullptr, nullptr, 0, 1);
      maxk_kernel<<<dim3((64 * NPT + 255) / 256), 256, 0, stream>>>(
          E2, h + (long long)b * sH + (long long)rowOff * NPT, 64);
    }
  };

  dim3 gN((NPT + 255) / 256, BATCH);

  // ---- split x into COR (6) and RGB (3) ----
  split_kernel<<<gN, 256, 0, stream>>>(P(I_X), COR, RGB);

  // ---- layer 1: knn on COR[:, :3]; edge conv 12->32->64 ----
  sqnorm_kernel<<<gN, 256, 0, stream>>>(COR, 6ll * NPT, 3, NRM);
  knn_kernel<<<dim3((NPT + 127) / 128, BATCH), 128, 0, stream>>>(COR, 6ll * NPT, 3, NRM, IDX);
  edge_layer(COR, 6, W1, BN1, 12, 32, W2, BN2, 1024);              // cor1 -> h[1024:1088]
  // rgb1: 3->32->64 (relu)
  gemm(P(W1D), 3, 0, 0, RGB, NPT, 3ll * NPT, 0, T64, NPT, 64ll * NPT,
       32, NPT, 3, BN1D, 1, nullptr, nullptr, 0, BATCH);
  gemm(P(W2D), 32, 0, 0, T64, NPT, 64ll * NPT, 0, h + 1088ll * NPT, NPT, sH,
       64, NPT, 32, BN2D, 1, nullptr, nullptr, 0, BATCH);          // rgb1 -> h[1088:1152]

  // ---- layer 2: knn on concat(COR, cor1) (70 ch); edge 140->64->64 ----
  copyR(CC,             70ll * NPT, COR,              6ll * NPT, 6);
  copyR(CC + 6ll * NPT, 70ll * NPT, h + 1024ll * NPT, sH,        64);
  sqnorm_kernel<<<gN, 256, 0, stream>>>(CC, 70ll * NPT, 70, NRM);
  knn_kernel<<<dim3((NPT + 127) / 128, BATCH), 128, 0, stream>>>(CC, 70ll * NPT, 70, NRM, IDX);
  edge_layer(CC, 70, W3, BN3, 140, 64, W4, BN4, 1216);             // cor2 -> h[1216:1280]
  // rgb2: concat(RGB, rgb1) (67) -> 64 -> 64
  copyR(RC,             67ll * NPT, RGB,              3ll * NPT, 3);
  copyR(RC + 3ll * NPT, 67ll * NPT, h + 1088ll * NPT, sH,        64);
  gemm(P(W3D), 67, 0, 0, RC, NPT, 67ll * NPT, 0, T64, NPT, 64ll * NPT,
       64, NPT, 67, BN3D, 1, nullptr, nullptr, 0, BATCH);
  gemm(P(W4D), 64, 0, 0, T64, NPT, 64ll * NPT, 0, h + 1152ll * NPT, NPT, sH,
       64, NPT, 64, BN4D, 1, nullptr, nullptr, 0, BATCH);          // rgb2 -> h[1152:1216]

  // ---- layer 3: knn on concat(COR, cor2, cor1) (134 ch); edge 268->64->64 ----
  copyR(CC,              134ll * NPT, COR,              6ll * NPT, 6);
  copyR(CC + 6ll  * NPT, 134ll * NPT, h + 1216ll * NPT, sH,        64);
  copyR(CC + 70ll * NPT, 134ll * NPT, h + 1024ll * NPT, sH,        64);
  sqnorm_kernel<<<gN, 256, 0, stream>>>(CC, 134ll * NPT, 134, NRM);
  knn_kernel<<<dim3((NPT + 127) / 128, BATCH), 128, 0, stream>>>(CC, 134ll * NPT, 134, NRM, IDX);
  edge_layer(CC, 134, W5, BN5, 268, 64, WC, BNC, 1344);            // cor3 -> h[1344:1408]
  // rgb3: concat(RGB, rgb2, rgb1) (131) -> 64 -> 64
  copyR(RC,              131ll * NPT, RGB,              3ll * NPT, 3);
  copyR(RC + 3ll  * NPT, 131ll * NPT, h + 1152ll * NPT, sH,        64);
  copyR(RC + 67ll * NPT, 131ll * NPT, h + 1088ll * NPT, sH,        64);
  gemm(P(W5D), 131, 0, 0, RC, NPT, 131ll * NPT, 0, T64, NPT, 64ll * NPT,
       64, NPT, 131, BN5D, 1, nullptr, nullptr, 0, BATCH);
  gemm(P(W6D), 64, 0, 0, T64, NPT, 64ll * NPT, 0, h + 1280ll * NPT, NPT, sH,
       64, NPT, 64, BN6D, 1, nullptr, nullptr, 0, BATCH);          // rgb3 -> h[1280:1344]

  // ---- global feature: g = max_n cbl(feats, w6); feats = h rows 1024..1407 ----
  gemm(P(W6), 384, 0, 0, h + 1024ll * NPT, NPT, sH, 0, gful, NPT, 1024ll * NPT,
       1024, NPT, 384, BN6, 2, nullptr, nullptr, 0, BATCH);
  maxbcast_kernel<<<dim3(1024, BATCH), 256, 0, stream>>>(gful, 1024ll * NPT, h, sH);

  // ---- CAA attention ----
  // q/kk[b] (P x C): A = wq/wk (P x N), B = h^T via transB
  gemm(P(WQ), NPT, 0, 0, h, NPT, sH, 1, qb,  CH, (long long)PDIM * CH,
       PDIM, CH, NPT, BNQ, 1, nullptr, nullptr, 0, BATCH);
  gemm(P(WK), NPT, 0, 0, h, NPT, sH, 1, kkb, CH, (long long)PDIM * CH,
       PDIM, CH, NPT, BNK, 1, nullptr, nullptr, 0, BATCH);
  // sim[b] (C x C) = kk^T @ q
  gemm(kkb, CH, (long long)PDIM * CH, 1, qb, CH, (long long)PDIM * CH, 0,
       simb, CH, (long long)CH * CH, CH, CH, PDIM, -1, 0, nullptr, nullptr, 0, BATCH);
  negsoftmax_kernel<<<dim3(BATCH * CH), 256, 0, stream>>>(simb, CH);
  // v[b] = relu(bn(wv @ h))
  gemm(P(WV), CH, 0, 0, h, NPT, sH, 0, vbuf, NPT, sH,
       CH, NPT, CH, BNV, 1, nullptr, nullptr, 0, BATCH);
  // h = alpha * (aff @ v) + h   (in-place residual)
  gemm(simb, CH, (long long)CH * CH, 0, vbuf, NPT, sH, 0, h, NPT, sH,
       CH, NPT, CH, -1, 0, P(I_ALPHA), h, sH, BATCH);

  // ---- head: 1408 -> 512 -> 256 -> 128 -> 13 ----
  gemm(P(W7), CH,  0, 0, h,  NPT, sH,          0, h7, NPT, 512ll * NPT,
       512, NPT, CH,  BN7, 2, nullptr, nullptr, 0, BATCH);
  gemm(P(W8), 512, 0, 0, h7, NPT, 512ll * NPT, 0, h8, NPT, 256ll * NPT,
       256, NPT, 512, BN8, 2, nullptr, nullptr, 0, BATCH);
  gemm(P(W9), 256, 0, 0, h8, NPT, 256ll * NPT, 0, h9, NPT, 128ll * NPT,
       128, NPT, 256, BN9, 2, nullptr, nullptr, 0, BATCH);
  gemm(P(W10), 128, 0, 0, h9, NPT, 128ll * NPT, 0, (float*)d_out, NPT, 13ll * NPT,
       13, NPT, 128, -1, 0, nullptr, nullptr, 0, BATCH);
}
